// Attn_88759794139141
// MI455X (gfx1250) — matrix operand order repaired and checked by reading, not compiled, not run on hardware
//
#include <hip/hip_runtime.h>
#include <hip/hip_bf16.h>

typedef __bf16 bf16_t;
typedef __attribute__((ext_vector_type(16))) __bf16 v16bf;
typedef __attribute__((ext_vector_type(8)))  __bf16 v8bf;
typedef __attribute__((ext_vector_type(8)))  float  v8f;
typedef __attribute__((ext_vector_type(4)))  float  v4f;

#define D_MODEL 1024
#define NHEAD   16
#define HDIM    64
#define BATCH   2
#define SEQ     2048
#define NTOK    (BATCH * SEQ)   // 4096

// ---------------------------------------------------------------------------
// WMMA wrapper: D = A(16x32 bf16) * B(32x16 bf16) + C(16x16 f32)
// ---------------------------------------------------------------------------
__device__ __forceinline__ v8f wmma_bf16(v16bf a, v16bf b, v8f c) {
  return __builtin_amdgcn_wmma_f32_16x16x32_bf16(
      /*neg_a=*/false, a, /*neg_b=*/false, b,
      /*c_mod=*/(short)0, c, /*reuse_a=*/false, /*reuse_b=*/false);
}

// ---------------------------------------------------------------------------
// Fragment layouts (wave32, CDNA5 ISA 7.12.2)
// A 16x32 (16-bit, contiguous k, not ISA table order): lane l<16 -> row l, elems 0..15 = K k0..k0+15
//                   lane l+16 -> row l, elems 0..15 = K k0+16..k0+31
// B 32x16 (16-bit, same contiguous k as A, not ISA table order): lane n<16 -> col n, elems 0..15 = K k0..k0+15
//                   lane n+16 -> col n, elems 0..15 = K k0+16..k0+31
// C/D 16x16 f32:    lane n<16 -> col n rows 0..7 ; lane n+16 -> col n rows 8..15
// ---------------------------------------------------------------------------
__device__ __forceinline__ v16bf frag_from(const bf16_t* p, int stride16) {
  // elems 0..7 at p[0..7], elems 8..15 at p[stride16 .. stride16+7]
  v8bf c0 = *(const v8bf*)(p);
  v8bf c1 = *(const v8bf*)(p + stride16);
  v16bf f;
#pragma unroll
  for (int i = 0; i < 8; ++i) { f[i] = c0[i]; f[i + 8] = c1[i]; }
  return f;
}

__device__ __forceinline__ v16bf load_a_gbl(const bf16_t* __restrict__ base,
                                            int ld, int row, int k0, int lane) {
  const int l = lane & 15, hi = lane >> 4;
  return frag_from(base + (size_t)(row + l) * ld + k0 + hi * 16, 8);
}

__device__ __forceinline__ v16bf load_b_gbl(const bf16_t* __restrict__ base,
                                            int ld, int col0, int k0, int lane) {
  const int n = lane & 15, hi = lane >> 4;
  return frag_from(base + (size_t)(col0 + n) * ld + k0 + hi * 16, 8);
}

// LDS fragment readers -------------------------------------------------------
__device__ __forceinline__ v16bf lds_a_k(const bf16_t (&T)[32][HDIM],
                                         int row0, int k0, int lane) {
  const int l = lane & 15, hi = lane >> 4;
  return frag_from(&T[row0 + l][k0 + hi * 16], 8);
}

__device__ __forceinline__ v16bf lds_a_v(const bf16_t (&T)[HDIM][32],
                                         int row0, int lane) {
  const int l = lane & 15, hi = lane >> 4;
  return frag_from(&T[row0 + l][hi * 16], 8);
}

__device__ __forceinline__ v16bf lds_b(const bf16_t (&T)[64][32],
                                       int col0, int lane) {
  const int n = lane & 15, hi = lane >> 4;
  return frag_from(&T[col0 + n][hi * 16], 8);
}

// ---------------------------------------------------------------------------
// Kernel 0: elementwise f32 -> bf16 (8 elems/thread, b128 in / b128 out)
// ---------------------------------------------------------------------------
__global__ __launch_bounds__(256) void cvt_kernel(const float* __restrict__ src,
                                                  bf16_t* __restrict__ dst) {
  const size_t i = ((size_t)blockIdx.x * 256 + threadIdx.x) * 8;
  v4f f0 = *(const v4f*)(src + i);
  v4f f1 = *(const v4f*)(src + i + 4);
  v8bf o;
#pragma unroll
  for (int j = 0; j < 4; ++j) { o[j] = (__bf16)f0[j]; o[j + 4] = (__bf16)f1[j]; }
  *(v8bf*)(dst + i) = o;
}

// ---------------------------------------------------------------------------
// Kernel 1: fused QKV projection.  C[4096,3072] = Xbf * Wqkv^T + bqkv
// Block = 8 waves, tile 256(M) x 64(N). Shared B tile (64 x 32 per chunk) is
// double-buffered in LDS (staged once per block instead of once per wave).
// Epilogue scatters to Q[b,h,s,d] (pre-scaled 1/8), K[b,h,s,d], Vt[b,h,d,s].
// ---------------------------------------------------------------------------
__global__ __launch_bounds__(256) void qkv_kernel(
    const bf16_t* __restrict__ Xbf, const bf16_t* __restrict__ Wq,
    const float* __restrict__ bqkv,
    bf16_t* __restrict__ Qbf, bf16_t* __restrict__ Kbf, bf16_t* __restrict__ Vt) {
  __shared__ alignas(16) bf16_t Bs[2][64][32];   // 8 KB
  const int lane = threadIdx.x & 31;
  const int wave = threadIdx.x >> 5;
  const int mbase = blockIdx.y * 256 + wave * 32;
  const int nbase = blockIdx.x * 64;

  const int srow = threadIdx.x >> 2;             // 0..63
  const int scol = (threadIdx.x & 3) * 8;        // 0,8,16,24

  // stage chunk 0
  *(v8bf*)&Bs[0][srow][scol] =
      *(const v8bf*)(Wq + (size_t)(nbase + srow) * D_MODEL + scol);

  v8f acc[2][4] = {};
  for (int kc = 0; kc < D_MODEL; kc += 32) {
    const int cur = (kc >> 5) & 1;
    __syncthreads();
    if (kc + 32 < D_MODEL) {
      *(v8bf*)&Bs[cur ^ 1][srow][scol] =
          *(const v8bf*)(Wq + (size_t)(nbase + srow) * D_MODEL + kc + 32 + scol);
    }
    v16bf a0 = load_a_gbl(Xbf, D_MODEL, mbase, kc, lane);
    v16bf a1 = load_a_gbl(Xbf, D_MODEL, mbase + 16, kc, lane);
#pragma unroll
    for (int t = 0; t < 4; ++t) {
      v16bf b = lds_b(Bs[cur], t * 16, lane);
      acc[0][t] = wmma_bf16(a0, b, acc[0][t]);
      acc[1][t] = wmma_bf16(a1, b, acc[1][t]);
    }
  }

  const int n = lane & 15, hi = lane >> 4;
#pragma unroll
  for (int u = 0; u < 2; ++u) {
#pragma unroll
    for (int t = 0; t < 4; ++t) {
#pragma unroll
      for (int r = 0; r < 8; ++r) {
        const int m = mbase + u * 16 + r + 8 * hi;   // token index 0..4095
        const int e = nbase + t * 16 + n;            // feature 0..3071
        const float val = acc[u][t][r] + bqkv[e];
        const int which = e >> 10;                   // 0=Q 1=K 2=V
        const int h  = (e >> 6) & (NHEAD - 1);
        const int dk = e & (HDIM - 1);
        const int bb = m >> 11;
        const int s  = m & (SEQ - 1);
        if (which == 0) {
          Qbf[(((size_t)(bb * NHEAD + h)) * SEQ + s) * HDIM + dk] =
              (__bf16)(val * 0.125f);                // fold 1/sqrt(64)
        } else if (which == 1) {
          Kbf[(((size_t)(bb * NHEAD + h)) * SEQ + s) * HDIM + dk] = (__bf16)val;
        } else {
          Vt[(((size_t)(bb * NHEAD + h)) * HDIM + dk) * SEQ + s] = (__bf16)val;
        }
      }
    }
  }
}

// ---------------------------------------------------------------------------
// Kernel 2: flash attention. Block = 8 waves = 128 queries of one (b,h).
// K/V^T tiles for each 32-key step are shared by all waves via double-buffered
// LDS (8x less global traffic). St = K*Q^T in C-layout -> per-column online
// softmax (one shfl_xor(16) per reduction); O^T accum with A = Vt rows.
// Grid = (S/128, B*H), block = 256.
// ---------------------------------------------------------------------------
__global__ __launch_bounds__(256) void attn_kernel(
    const bf16_t* __restrict__ Qbf, const bf16_t* __restrict__ Kbf,
    const bf16_t* __restrict__ Vt, bf16_t* __restrict__ Obf) {
  __shared__ alignas(16) bf16_t Ks[2][32][HDIM];   // 8 KB
  __shared__ alignas(16) bf16_t Vs[2][HDIM][32];   // 8 KB
  const int lane = threadIdx.x & 31;
  const int wave = threadIdx.x >> 5;
  const int bh = blockIdx.y;                       // 0..31
  const int qbase = blockIdx.x * 128 + wave * 16;
  const int hi = lane >> 4;

  const bf16_t* Qh = Qbf + (size_t)bh * SEQ * HDIM;
  const bf16_t* Kh = Kbf + (size_t)bh * SEQ * HDIM;
  const bf16_t* Vh = Vt + (size_t)bh * HDIM * SEQ;

  // Q as B-fragments (queries = columns), head-dim split 0..31 / 32..63
  const v16bf q0 = load_b_gbl(Qh, HDIM, qbase, 0, lane);
  const v16bf q1 = load_b_gbl(Qh, HDIM, qbase, 32, lane);

  const int krow = threadIdx.x >> 3, kcol = (threadIdx.x & 7) * 8;  // 32x8 chunks
  const int vrow = threadIdx.x >> 2, vcol = (threadIdx.x & 3) * 8;  // 64x4 chunks

  // stage key-block 0
  *(v8bf*)&Ks[0][krow][kcol] = *(const v8bf*)(Kh + (size_t)krow * HDIM + kcol);
  *(v8bf*)&Vs[0][vrow][vcol] = *(const v8bf*)(Vh + (size_t)vrow * SEQ + vcol);

  v8f accO[4] = {};                                // O^T tiles: d rows x q cols
  float m_run = -__builtin_inff();
  float l_run = 0.0f;

  for (int kb = 0; kb < SEQ; kb += 32) {
    const int cur = (kb >> 5) & 1;
    __syncthreads();
    if (kb + 32 < SEQ) {
      *(v8bf*)&Ks[cur ^ 1][krow][kcol] =
          *(const v8bf*)(Kh + (size_t)(kb + 32 + krow) * HDIM + kcol);
      *(v8bf*)&Vs[cur ^ 1][vrow][vcol] =
          *(const v8bf*)(Vh + (size_t)vrow * SEQ + kb + 32 + vcol);
    }

    // Scores^T for 32 keys x 16 queries (two 16x16 tiles)
    v16bf ak;
    v8f st0 = {}, st1 = {};
    ak = lds_a_k(Ks[cur], 0, 0, lane);   st0 = wmma_bf16(ak, q0, st0);
    ak = lds_a_k(Ks[cur], 0, 32, lane);  st0 = wmma_bf16(ak, q1, st0);
    ak = lds_a_k(Ks[cur], 16, 0, lane);  st1 = wmma_bf16(ak, q0, st1);
    ak = lds_a_k(Ks[cur], 16, 32, lane); st1 = wmma_bf16(ak, q1, st1);

    // ---- online softmax (per query column; lanes n and n+16 share a query)
    float mb = -__builtin_inff();
#pragma unroll
    for (int r = 0; r < 8; ++r) mb = fmaxf(mb, fmaxf(st0[r], st1[r]));
    mb = fmaxf(mb, __shfl_xor(mb, 16, 32));
    const float m_new = fmaxf(m_run, mb);
    const float corr = __expf(m_run - m_new);

    float p0[8], p1[8], ssum = 0.0f;
#pragma unroll
    for (int r = 0; r < 8; ++r) {
      p0[r] = __expf(st0[r] - m_new);
      p1[r] = __expf(st1[r] - m_new);
      ssum += p0[r] + p1[r];
    }
    ssum += __shfl_xor(ssum, 16, 32);
    l_run = l_run * corr + ssum;
    m_run = m_new;

    // ---- build P as B-fragment (contraction over 32 keys, cols = queries):
    // lane n<16 needs keys kb..kb+15  = own st0 (0..7) + partner st0 (8..15)
    // lane n+16 needs keys kb+16..+31 = partner st1 (0..7) + own st1 (8..15)
    v16bf p;
#pragma unroll
    for (int r = 0; r < 8; ++r) {
      const float o0 = __shfl_xor(p0[r], 16, 32);
      const float o1 = __shfl_xor(p1[r], 16, 32);
      p[r]     = (__bf16)(hi ? o1 : p0[r]);
      p[r + 8] = (__bf16)(hi ? p1[r] : o0);
    }

    // ---- accO = accO*corr + Vt_tile * P
#pragma unroll
    for (int t = 0; t < 4; ++t) {
#pragma unroll
      for (int r = 0; r < 8; ++r) accO[t][r] *= corr;
      v16bf av = lds_a_v(Vs[cur], t * 16, lane);
      accO[t] = wmma_bf16(av, p, accO[t]);
    }
  }

  // ---- epilogue: O[b, s, h*64 + d] = accO(d, q) / l
  const float inv_l = 1.0f / l_run;
  const int b = bh >> 4, h = bh & (NHEAD - 1);
  const int n = lane & 15;
#pragma unroll
  for (int t = 0; t < 4; ++t) {
#pragma unroll
    for (int r = 0; r < 8; ++r) {
      const int d = t * 16 + r + 8 * hi;
      const size_t idx =
          ((size_t)(b * SEQ + qbase + n)) * D_MODEL + h * HDIM + d;
      Obf[idx] = (__bf16)(accO[t][r] * inv_l);
    }
  }
}

// ---------------------------------------------------------------------------
// Kernel 3: output projection. out[4096,1024] = Obf * Wproj^T + bproj (f32)
// Same LDS-staged structure as kernel 1. Grid = (1024/64, 4096/256).
// ---------------------------------------------------------------------------
__global__ __launch_bounds__(256) void proj_kernel(
    const bf16_t* __restrict__ Obf, const bf16_t* __restrict__ Wp,
    const float* __restrict__ bproj, float* __restrict__ out) {
  __shared__ alignas(16) bf16_t Bs[2][64][32];   // 8 KB
  const int lane = threadIdx.x & 31;
  const int wave = threadIdx.x >> 5;
  const int mbase = blockIdx.y * 256 + wave * 32;
  const int nbase = blockIdx.x * 64;

  const int srow = threadIdx.x >> 2;
  const int scol = (threadIdx.x & 3) * 8;

  *(v8bf*)&Bs[0][srow][scol] =
      *(const v8bf*)(Wp + (size_t)(nbase + srow) * D_MODEL + scol);

  v8f acc[2][4] = {};
  for (int kc = 0; kc < D_MODEL; kc += 32) {
    const int cur = (kc >> 5) & 1;
    __syncthreads();
    if (kc + 32 < D_MODEL) {
      *(v8bf*)&Bs[cur ^ 1][srow][scol] =
          *(const v8bf*)(Wp + (size_t)(nbase + srow) * D_MODEL + kc + 32 + scol);
    }
    v16bf a0 = load_a_gbl(Obf, D_MODEL, mbase, kc, lane);
    v16bf a1 = load_a_gbl(Obf, D_MODEL, mbase + 16, kc, lane);
#pragma unroll
    for (int t = 0; t < 4; ++t) {
      v16bf b = lds_b(Bs[cur], t * 16, lane);
      acc[0][t] = wmma_bf16(a0, b, acc[0][t]);
      acc[1][t] = wmma_bf16(a1, b, acc[1][t]);
    }
  }

  const int n = lane & 15, hi = lane >> 4;
#pragma unroll
  for (int u = 0; u < 2; ++u) {
#pragma unroll
    for (int t = 0; t < 4; ++t) {
#pragma unroll
      for (int r = 0; r < 8; ++r) {
        const int m = mbase + u * 16 + r + 8 * hi;
        const int e = nbase + t * 16 + n;
        out[(size_t)m * D_MODEL + e] = acc[u][t][r] + bproj[e];
      }
    }
  }
}

// ---------------------------------------------------------------------------
extern "C" void kernel_launch(void* const* d_in, const int* in_sizes, int n_in,
                              void* d_out, int out_size, void* d_ws,
                              size_t ws_size, hipStream_t stream) {
  const float* x     = (const float*)d_in[0];
  const float* Wqkv  = (const float*)d_in[1];
  const float* bqkv  = (const float*)d_in[2];
  const float* Wproj = (const float*)d_in[3];
  const float* bproj = (const float*)d_in[4];
  float* out = (float*)d_out;

  char* ws = (char*)d_ws;
  const size_t seg = (size_t)NTOK * D_MODEL * sizeof(bf16_t);  // 8 MB
  bf16_t* Qbf  = (bf16_t*)(ws + 0 * seg);
  bf16_t* Kbf  = (bf16_t*)(ws + 1 * seg);
  bf16_t* Vt   = (bf16_t*)(ws + 2 * seg);
  bf16_t* Obf  = (bf16_t*)(ws + 3 * seg);
  bf16_t* Xbf  = (bf16_t*)(ws + 4 * seg);
  bf16_t* Wqbf = (bf16_t*)(ws + 5 * seg);                      // 6 MB
  bf16_t* Wpbf = (bf16_t*)(ws + 5 * seg +
                           (size_t)3 * D_MODEL * D_MODEL * sizeof(bf16_t));

  dim3 blk(256);
  const int nX  = NTOK * D_MODEL;            // 4M elems
  const int nWq = 3 * D_MODEL * D_MODEL;     // 3M elems
  const int nWp = D_MODEL * D_MODEL;         // 1M elems
  cvt_kernel<<<dim3(nX  / (256 * 8)), blk, 0, stream>>>(x, Xbf);
  cvt_kernel<<<dim3(nWq / (256 * 8)), blk, 0, stream>>>(Wqkv, Wqbf);
  cvt_kernel<<<dim3(nWp / (256 * 8)), blk, 0, stream>>>(Wproj, Wpbf);

  qkv_kernel<<<dim3(3 * D_MODEL / 64, NTOK / 256), blk, 0, stream>>>(
      Xbf, Wqbf, bqkv, Qbf, Kbf, Vt);
  attn_kernel<<<dim3(SEQ / 128, BATCH * NHEAD), blk, 0, stream>>>(
      Qbf, Kbf, Vt, Obf);
  proj_kernel<<<dim3(D_MODEL / 64, NTOK / 256), blk, 0, stream>>>(
      Obf, Wpbf, bproj, out);
}
